// MinLSTMCell_76072460747249
// MI455X (gfx1250) — matrix operand order repaired and checked by reading, not compiled, not run on hardware
//
#include <hip/hip_runtime.h>
#include <hip/hip_bf16.h>
#include <math.h>

// ---------------------------------------------------------------------------
// MinLSTM-with-LayerNorm cell for MI455X (gfx1250, wave32).
//   1) f32 -> bf16 conversion of x [16M] and W [4M]
//   2) bf16 WMMA GEMM (M=16384,N=4096,K=1024), tiles staged via
//      GLOBAL_LOAD_ASYNC_TO_LDS_B128 (ASYNCcnt / s_wait_asynccnt), fragments
//      read with ds_load_b128, fused bias+sigmoid/tanh epilogue
//   3) chunked parallel scan of c_t = f_t*c + i_t*z_t (16 chunks x 128 steps)
//   4) per-row LayerNorm + tanh + o* epilogue -> h [B,S,H]
// ---------------------------------------------------------------------------

typedef __attribute__((ext_vector_type(16))) __bf16 v16bf;
typedef __attribute__((ext_vector_type(8)))  float  v8f;

union FragU { uint4 u[2]; v16bf v; };

// ----------------------------- f32 -> bf16 (RNE) ---------------------------
__global__ void cvt_bf16(const float* __restrict__ in,
                         unsigned short* __restrict__ out, int n) {
  int i = blockIdx.x * blockDim.x + threadIdx.x;
  if (i < n) {
    unsigned u = __float_as_uint(in[i]);
    unsigned r = u + 0x7FFFu + ((u >> 16) & 1u);   // round-to-nearest-even
    out[i] = (unsigned short)(r >> 16);
  }
}

// --------------------- WMMA GEMM + bias + gate activations -----------------
// gates layout: [row=b*S+s][4096], f:[0,1K) i:[1K,2K) o:[2K,3K) z:[3K,4K)
#define LDS_ROW 40   // 32 bf16 of data + 4-dword pad (80B row; stride-20 banks)

__global__ __launch_bounds__(256)
void gemm_gates(const unsigned short* __restrict__ xb,   // [16384,1024] bf16
                const unsigned short* __restrict__ wb,   // [4096,1024]  bf16
                const float* __restrict__ bias,          // [4096]
                float* __restrict__ gates)               // [16384,4096] f32
{
  const int K = 1024, N4 = 4096;
  // Double-buffered LDS tiles: A 128x32 bf16, B 64x32 bf16 (padded rows)
  __shared__ __align__(16) unsigned short smA[2][128 * LDS_ROW];
  __shared__ __align__(16) unsigned short smB[2][64 * LDS_ROW];

  const int tid  = threadIdx.x;
  const int lane = tid & 31, wave = tid >> 5;
  const int waveM = wave & 3;            // 4 M sub-tiles of 32 rows
  const int waveN = wave >> 2;           // 2 N sub-tiles of 32 cols
  const int half  = lane >> 4;           // 0 | 1
  const int r     = lane & 15;

  const int mBlock = blockIdx.y * 128;
  const int nBlock = blockIdx.x * 64;

  // Async-copy work split: A tile = 512 x 16B segs (2/thread), B = 256 (1/thread)
  const int aRow = tid >> 1;             // 0..127
  const int aK16 = (tid & 1) * 2;        // segment pair 0-1 or 2-3 within row
  const int bRow = tid >> 2;             // 0..63
  const int bK16 = tid & 3;              // 0..3

  const unsigned short* aG = xb + (size_t)(mBlock + aRow) * K + aK16 * 8;
  const unsigned short* bG = wb + (size_t)(nBlock + bRow) * K + bK16 * 8;
  const unsigned aL[2] = {
    (unsigned)(uintptr_t)&smA[0][aRow * LDS_ROW + aK16 * 8],
    (unsigned)(uintptr_t)&smA[1][aRow * LDS_ROW + aK16 * 8] };
  const unsigned bL[2] = {
    (unsigned)(uintptr_t)&smB[0][bRow * LDS_ROW + bK16 * 8],
    (unsigned)(uintptr_t)&smB[1][bRow * LDS_ROW + bK16 * 8] };

  // Issue the 3 async global->LDS b128 copies for one K-step into buffer `buf`
  auto issue = [&](int buf, int kb) {
    unsigned long long ga0 = (unsigned long long)(uintptr_t)(aG + kb);
    unsigned long long ga1 = ga0 + 16ull;                  // second A segment
    unsigned long long gb0 = (unsigned long long)(uintptr_t)(bG + kb);
    asm volatile("global_load_async_to_lds_b128 %0, %1, off"
                 :: "v"(aL[buf]),        "v"(ga0) : "memory");
    asm volatile("global_load_async_to_lds_b128 %0, %1, off"
                 :: "v"(aL[buf] + 16u),  "v"(ga1) : "memory");
    asm volatile("global_load_async_to_lds_b128 %0, %1, off"
                 :: "v"(bL[buf]),        "v"(gb0) : "memory");
    // Warm GL2 for the K-step after next (speculative, silently dropped OOB)
    __builtin_prefetch(aG + kb + 64, 0, 0);
    __builtin_prefetch(bG + kb + 64, 0, 0);
  };

  const v8f vzero = {0.f, 0.f, 0.f, 0.f, 0.f, 0.f, 0.f, 0.f};
  v8f acc[2][2];
  acc[0][0] = vzero; acc[0][1] = vzero; acc[1][0] = vzero; acc[1][1] = vzero;

  issue(0, 0);                                   // prologue: fill buffer 0
  int cur = 0;
  for (int kb = 0; kb < K; kb += 32) {
    const bool more = (kb + 32) < K;             // uniform
    if (more) {
      issue(cur ^ 1, kb + 32);                   // prefetch next buffer
      asm volatile("s_wait_asynccnt 0x3" ::: "memory");  // cur's 3 ops done
    } else {
      asm volatile("s_wait_asynccnt 0x0" ::: "memory");
    }
    __syncthreads();                             // cur tile visible to all

    // Fragments from LDS (ds_load_b128), WMMA layouts:
    //  A 16x32: half0 -> K{0..7,16..23}, half1 -> K{8..15,24..31}
    //  B 32x16: lane = column, half0 -> K{0..7,16..23}, half1 -> K{8..15,24..31}
    FragU a[2], b[2];
#pragma unroll
    for (int mt = 0; mt < 2; ++mt) {
      const unsigned short* aF =
          &smA[cur][(waveM * 32 + mt * 16 + r) * LDS_ROW];
      a[mt].u[0] = *(const uint4*)(aF + half * 8);
      a[mt].u[1] = *(const uint4*)(aF + 16 + half * 8);
    }
#pragma unroll
    for (int nt = 0; nt < 2; ++nt) {
      const unsigned short* bF =
          &smB[cur][(waveN * 32 + nt * 16 + r) * LDS_ROW];
      b[nt].u[0] = *(const uint4*)(bF + half * 8);
      b[nt].u[1] = *(const uint4*)(bF + 16 + half * 8);
    }
#pragma unroll
    for (int mt = 0; mt < 2; ++mt)
#pragma unroll
      for (int nt = 0; nt < 2; ++nt)
        acc[mt][nt] = __builtin_amdgcn_wmma_f32_16x16x32_bf16(
            false, a[mt].v, false, b[nt].v, (short)0, acc[mt][nt], false, false);

    __syncthreads();      // all reads of `cur` done before it is re-issued
    cur ^= 1;
  }

  // Epilogue: C layout -> row = base + mt*16 + 8*half + j, col = base + nt*16 + r
#pragma unroll
  for (int nt = 0; nt < 2; ++nt) {
    const int col = nBlock + waveN * 32 + nt * 16 + r;
    const float bv = bias[col];
    const int gate = col >> 10;                 // 0=f 1=i 2=o 3=z
#pragma unroll
    for (int mt = 0; mt < 2; ++mt) {
#pragma unroll
      for (int j = 0; j < 8; ++j) {
        const int row = mBlock + waveM * 32 + mt * 16 + 8 * half + j;
        float v = acc[mt][nt][j] + bv;
        v = (gate == 3) ? tanhf(v) : (1.f / (1.f + __expf(-v)));
        gates[(size_t)row * N4 + col] = v;
      }
    }
  }
}

// --------------------- chunked parallel scan over S ------------------------
// S=2048 in 16 chunks of 128. BH = B*H = 8192. tid = chunk*BH + (b*1024+h).
__global__ void scan_phase1(const float* __restrict__ gates,
                            float* __restrict__ Aarr, float* __restrict__ Barr) {
  const int tid = blockIdx.x * blockDim.x + threadIdx.x;     // 0..131071
  const int chunk = tid >> 13, idx = tid & 8191;
  const int b = idx >> 10, h = idx & 1023;
  size_t base = ((size_t)b * 2048 + (size_t)chunk * 128) * 4096 + h;
  float A = 1.f, Bv = 0.f;
  for (int t = 0; t < 128; ++t) {
    const float f = gates[base];
    const float i = gates[base + 1024];
    const float z = gates[base + 3072];
    A  = A * f;
    Bv = f * Bv + i * z;
    base += 4096;
  }
  Aarr[tid] = A; Barr[tid] = Bv;
}

__global__ void scan_phase2(const float* __restrict__ Aarr,
                            const float* __restrict__ Barr,
                            float* __restrict__ carry) {
  const int idx = blockIdx.x * blockDim.x + threadIdx.x;     // 0..8191
  float c = 0.f;
#pragma unroll
  for (int ch = 0; ch < 16; ++ch) {
    carry[ch * 8192 + idx] = c;                              // c entering chunk
    c = Aarr[ch * 8192 + idx] * c + Barr[ch * 8192 + idx];
  }
}

__global__ void scan_phase3(const float* __restrict__ gates,
                            const float* __restrict__ carry,
                            float* __restrict__ cbuf) {      // [16384,1024]
  const int tid = blockIdx.x * blockDim.x + threadIdx.x;
  const int chunk = tid >> 13, idx = tid & 8191;
  const int b = idx >> 10, h = idx & 1023;
  size_t base = ((size_t)b * 2048 + (size_t)chunk * 128) * 4096 + h;
  size_t crow = ((size_t)b * 2048 + (size_t)chunk * 128) * 1024 + h;
  float c = carry[tid];
  for (int t = 0; t < 128; ++t) {
    const float f = gates[base];
    const float i = gates[base + 1024];
    const float z = gates[base + 3072];
    c = f * c + i * z;
    cbuf[crow] = c;
    base += 4096; crow += 1024;
  }
}

// --------------------- LayerNorm + tanh + o* -------------------------------
__global__ __launch_bounds__(256)
void ln_out(const float* __restrict__ cbuf, const float* __restrict__ gates,
            const float* __restrict__ lnw, const float* __restrict__ lnb,
            float* __restrict__ out) {
  const int row = blockIdx.x;                 // 0..16383
  const int tid = threadIdx.x;
  float cv[4], sum = 0.f, sq = 0.f;
#pragma unroll
  for (int k = 0; k < 4; ++k) {
    const int h = tid + k * 256;
    const float c = cbuf[(size_t)row * 1024 + h];
    cv[k] = c; sum += c; sq += c * c;
  }
#pragma unroll
  for (int off = 16; off > 0; off >>= 1) {    // wave32 reduction
    sum += __shfl_xor(sum, off, 32);
    sq  += __shfl_xor(sq,  off, 32);
  }
  __shared__ float sS[8], sQ[8];
  if ((tid & 31) == 0) { sS[tid >> 5] = sum; sQ[tid >> 5] = sq; }
  __syncthreads();
  sum = 0.f; sq = 0.f;
#pragma unroll
  for (int w = 0; w < 8; ++w) { sum += sS[w]; sq += sQ[w]; }
  const float mean = sum * (1.f / 1024.f);
  const float var  = sq  * (1.f / 1024.f) - mean * mean;
  const float rstd = rsqrtf(var + 1e-5f);
#pragma unroll
  for (int k = 0; k < 4; ++k) {
    const int h = tid + k * 256;
    const float o = gates[(size_t)row * 4096 + 2048 + h];
    const float y = (cv[k] - mean) * rstd * lnw[h] + lnb[h];
    out[(size_t)row * 1024 + h] = o * tanhf(y);
  }
}

// ---------------------------------------------------------------------------
extern "C" void kernel_launch(void* const* d_in, const int* in_sizes, int n_in,
                              void* d_out, int out_size, void* d_ws, size_t ws_size,
                              hipStream_t stream) {
  const float* x    = (const float*)d_in[0];   // [8,2048,1024]
  const float* W    = (const float*)d_in[1];   // [4096,1024]
  const float* bias = (const float*)d_in[2];   // [4096]
  const float* lnw  = (const float*)d_in[3];   // [1024]
  const float* lnb  = (const float*)d_in[4];   // [1024]
  float* out = (float*)d_out;                  // [8,2048,1024]

  char* ws = (char*)d_ws;
  size_t off = 0;
  unsigned short* xb = (unsigned short*)(ws + off); off += (size_t)16777216 * 2; // 32 MB
  unsigned short* wb = (unsigned short*)(ws + off); off += (size_t)4194304 * 2;  //  8 MB
  float* gates = (float*)(ws + off); off += (size_t)16384 * 4096 * 4;            // 256 MB
  float* cbuf  = (float*)(ws + off); off += (size_t)16384 * 1024 * 4;            //  64 MB
  float* Aarr  = (float*)(ws + off); off += (size_t)131072 * 4;
  float* Barr  = (float*)(ws + off); off += (size_t)131072 * 4;
  float* carry = (float*)(ws + off); off += (size_t)131072 * 4;                  // ~380 MB total

  cvt_bf16<<<16777216 / 256, 256, 0, stream>>>(x, xb, 16777216);
  cvt_bf16<<<4194304 / 256, 256, 0, stream>>>(W, wb, 4194304);

  dim3 grid(4096 / 64, 16384 / 128);           // (N tiles, M tiles)
  gemm_gates<<<grid, 256, 0, stream>>>(xb, wb, bias, gates);

  scan_phase1<<<131072 / 256, 256, 0, stream>>>(gates, Aarr, Barr);
  scan_phase2<<<8192 / 256,   256, 0, stream>>>(Aarr, Barr, carry);
  scan_phase3<<<131072 / 256, 256, 0, stream>>>(gates, carry, cbuf);

  ln_out<<<16384, 256, 0, stream>>>(cbuf, gates, lnw, lnb, out);
}
